// ICLTransformerLayer_4733053960676
// MI455X (gfx1250) — matrix operand order repaired and checked by reading, not compiled, not run on hardware
//
#include <hip/hip_runtime.h>

// ---------------------------------------------------------------------------
// Problem constants (from reference)
// ---------------------------------------------------------------------------
#define BB   8
#define NN   1024
#define DD   1024
#define HH   8
#define HDIM 128
#define DFF  4096
#define MTOT (BB * NN)       // 8192 rows of activations

typedef unsigned short ushort;
typedef __bf16 bf16;
typedef bf16  bf16x8  __attribute__((ext_vector_type(8)));
typedef bf16  bf16x16 __attribute__((ext_vector_type(16)));
typedef float floatx8 __attribute__((ext_vector_type(8)));

union AFrag { bf16x16 v; bf16x8 h[2]; };

__device__ __forceinline__ ushort f2bf(float f) {
    union { float f; unsigned u; } v; v.f = f;
    unsigned r = v.u + 0x7FFFu + ((v.u >> 16) & 1u);   // round-to-nearest-even
    return (ushort)(r >> 16);
}

// gfx1250 async global->LDS copy (ASYNCcnt-tracked, bypasses VGPRs).
// ldsaddr = LDS byte address (generic shared addr truncated to 32 bits,
// per ISA the LDS address is the low 32 address bits).
__device__ __forceinline__ void async_load_b128(unsigned ldsaddr,
                                                unsigned long long gaddr) {
    asm volatile("global_load_async_to_lds_b128 %0, %1, off"
                 :: "v"(ldsaddr), "v"(gaddr) : "memory");
}
__device__ __forceinline__ void wait_asynccnt0() {
#if __has_builtin(__builtin_amdgcn_s_wait_asynccnt)
    __builtin_amdgcn_s_wait_asynccnt(0);
#else
    asm volatile("s_wait_asynccnt 0x0" ::: "memory");
#endif
}

// ---------------------------------------------------------------------------
// fp32 -> bf16 elementwise convert
// ---------------------------------------------------------------------------
__global__ __launch_bounds__(256)
void conv_bf16_kernel(const float* __restrict__ in, ushort* __restrict__ out, int n) {
    int i = blockIdx.x * 256 + threadIdx.x;
    if (i < n) out[i] = f2bf(in[i]);
}

// ---------------------------------------------------------------------------
// fp32 [K][N] -> bf16 transposed [N][K]
// ---------------------------------------------------------------------------
__global__ __launch_bounds__(256)
void transpose_bf16_kernel(const float* __restrict__ W, ushort* __restrict__ Wt,
                           int K, int N) {
    __shared__ float tile[32][33];
    int kx = blockIdx.x * 32, nx = blockIdx.y * 32;
    int tx = threadIdx.x & 31, ty = threadIdx.x >> 5;   // 8 rows per pass
    #pragma unroll
    for (int i = 0; i < 32; i += 8)
        tile[ty + i][tx] = W[(size_t)(kx + ty + i) * N + nx + tx];
    __syncthreads();
    #pragma unroll
    for (int i = 0; i < 32; i += 8)
        Wt[(size_t)(nx + ty + i) * K + kx + tx] = f2bf(tile[tx][ty + i]);
}

// ---------------------------------------------------------------------------
// bf16 WMMA GEMM:  C[M][Nc] = A[M][K] * Bt[Nc][K]^T + bias, optional ReLU.
// Block = 256 threads (8 waves), tile 128x128, K-step 32.
// Double-buffered LDS filled with async global->LDS loads (ASYNCcnt),
// overlapping the next tile's fill with the current tile's WMMAs.
// Each wave: 2 (M) x 4 (N) tiles of v_wmma_f32_16x16x32_bf16.
// ---------------------------------------------------------------------------
template <bool RELU, bool OUTBF>
__global__ __launch_bounds__(256)
void gemm_bf16_kernel(const ushort* __restrict__ A,   // [M][K] bf16
                      const ushort* __restrict__ Bt,  // [Nc][K] bf16 (B transposed)
                      const float*  __restrict__ bias,// [Nc]
                      void* __restrict__ Cout, int Nc, int K) {
    __shared__ __align__(16) ushort sA[2][128 * 32];
    __shared__ __align__(16) ushort sB[2][128 * 32];

    const int tid  = threadIdx.x;
    const int wave = tid >> 5, lane = tid & 31;
    const int half = lane >> 4, l16 = lane & 15;
    const int bm = blockIdx.x * 128, bn = blockIdx.y * 128;
    const int wr = wave >> 1, wc = wave & 1;            // 4x2 wave grid

    const unsigned sAbase = (unsigned)(size_t)(const void*)&sA[0][0];
    const unsigned sBbase = (unsigned)(size_t)(const void*)&sB[0][0];

    // this thread's two 8-element chunks of the 128x32 tile
    const int row0 = tid >> 2,            col0 = (tid & 3) * 8;
    const int row1 = (tid + 256) >> 2,    col1 = ((tid + 256) & 3) * 8;

    floatx8 acc[2][4] = {};

    // prologue: fill buffer 0 for k0 = 0
    {
        async_load_b128(sAbase + (unsigned)(row0 * 32 + col0) * 2,
                        (unsigned long long)&A[(size_t)(bm + row0) * K + col0]);
        async_load_b128(sAbase + (unsigned)(row1 * 32 + col1) * 2,
                        (unsigned long long)&A[(size_t)(bm + row1) * K + col1]);
        async_load_b128(sBbase + (unsigned)(row0 * 32 + col0) * 2,
                        (unsigned long long)&Bt[(size_t)(bn + row0) * K + col0]);
        async_load_b128(sBbase + (unsigned)(row1 * 32 + col1) * 2,
                        (unsigned long long)&Bt[(size_t)(bn + row1) * K + col1]);
    }

    for (int k0 = 0, st = 0; k0 < K; k0 += 32, st++) {
        wait_asynccnt0();            // this wave's stage-st fills have landed
        __syncthreads();             // all waves' fills landed; other buffer free

        const int cur = st & 1;
        if (k0 + 32 < K) {           // overlap next tile's fill with compute
            const int nxt = cur ^ 1;
            const unsigned abuf = sAbase + (unsigned)(nxt * 128 * 32) * 2;
            const unsigned bbuf = sBbase + (unsigned)(nxt * 128 * 32) * 2;
            async_load_b128(abuf + (unsigned)(row0 * 32 + col0) * 2,
                (unsigned long long)&A[(size_t)(bm + row0) * K + k0 + 32 + col0]);
            async_load_b128(abuf + (unsigned)(row1 * 32 + col1) * 2,
                (unsigned long long)&A[(size_t)(bm + row1) * K + k0 + 32 + col1]);
            async_load_b128(bbuf + (unsigned)(row0 * 32 + col0) * 2,
                (unsigned long long)&Bt[(size_t)(bn + row0) * K + k0 + 32 + col0]);
            async_load_b128(bbuf + (unsigned)(row1 * 32 + col1) * 2,
                (unsigned long long)&Bt[(size_t)(bn + row1) * K + k0 + 32 + col1]);
        }

        #pragma unroll
        for (int mi = 0; mi < 2; mi++) {
            AFrag af;                                    // A 16x32 fragment
            const ushort* ap = &sA[cur][(wr * 32 + mi * 16 + l16) * 32];
            af.h[0] = *(const bf16x8*)&ap[half * 8];
            af.h[1] = *(const bf16x8*)&ap[16 + half * 8];
            #pragma unroll
            for (int ni = 0; ni < 4; ni++) {
                AFrag bf_;                               // B 32x16 fragment (from Bt)
                const ushort* bp = &sB[cur][(wc * 64 + ni * 16 + l16) * 32];
                bf_.h[0] = *(const bf16x8*)&bp[half * 8];
                bf_.h[1] = *(const bf16x8*)&bp[16 + half * 8];
                acc[mi][ni] = __builtin_amdgcn_wmma_f32_16x16x32_bf16(
                    false, af.v, false, bf_.v, (short)0, acc[mi][ni], false, false);
            }
        }
    }

    #pragma unroll
    for (int mi = 0; mi < 2; mi++)
        #pragma unroll
        for (int ni = 0; ni < 4; ni++) {
            int col = bn + wc * 64 + ni * 16 + l16;
            float bv = bias[col];
            #pragma unroll
            for (int r = 0; r < 8; r++) {
                int row = bm + wr * 32 + mi * 16 + r + 8 * half;
                float v = acc[mi][ni][r] + bv;
                if (RELU) v = v > 0.f ? v : 0.f;
                if (OUTBF) ((ushort*)Cout)[(size_t)row * Nc + col] = f2bf(v);
                else       ((float*)Cout)[(size_t)row * Nc + col] = v;
            }
        }
}

// ---------------------------------------------------------------------------
// Flash-style masked attention for one (b,h), 64 query rows per block.
// Block = 128 threads (4 waves); each wave owns 16 query rows.
// K/V tiles of 64 keys staged in LDS (V stored transposed for B-fragments).
// Online softmax with cross-lane reductions inside 16-lane halves.
// ---------------------------------------------------------------------------
__global__ __launch_bounds__(128)
void attn_kernel(const ushort* __restrict__ Qb, const ushort* __restrict__ Kb,
                 const ushort* __restrict__ Vb, const unsigned char* __restrict__ tgt,
                 ushort* __restrict__ Ob) {
    const int bh = blockIdx.x;
    const int b = bh / HH, h = bh % HH;
    const int qbase = blockIdx.y * 64;

    __shared__ __align__(16) ushort sK[64 * 128];      // [key][hd]
    __shared__ __align__(16) ushort sV[128 * 64];      // [hd][key] (transposed)
    __shared__ __align__(16) ushort sP[4][16 * 64];    // per-wave P tile

    const int tid  = threadIdx.x;
    const int wave = tid >> 5, lane = tid & 31;
    const int half = lane >> 4, l16 = lane & 15;
    const int q0   = qbase + wave * 16;
    const float inv_scale = 0.08838834764831845f;       // 1/sqrt(128)

    // Q fragments for the 4 K-steps over HD=128 (A-matrix layout, from global)
    AFrag qa[4];
    {
        const ushort* qp = &Qb[((size_t)b * NN + q0 + l16) * DD + h * HDIM];
        #pragma unroll
        for (int kk = 0; kk < 4; kk++) {
            qa[kk].h[0] = *(const bf16x8*)&qp[kk * 32 + half * 8];
            qa[kk].h[1] = *(const bf16x8*)&qp[kk * 32 + 16 + half * 8];
        }
    }
    // per-row target flags for this lane's 8 rows (row M = r + 8*half)
    bool tr[8];
    #pragma unroll
    for (int r = 0; r < 8; r++)
        tr[r] = tgt[(size_t)b * NN + q0 + r + 8 * half] != 0;

    floatx8 m, l, o[8];
    #pragma unroll
    for (int r = 0; r < 8; r++) { m[r] = -3.0e38f; l[r] = 0.f; }
    #pragma unroll
    for (int t = 0; t < 8; t++)
        #pragma unroll
        for (int r = 0; r < 8; r++) o[t][r] = 0.f;

    for (int j = 0; j < NN; j += 64) {
        __syncthreads();
        // stage K tile [64 keys][128 hd] and V tile transposed [128 hd][64 keys]
        #pragma unroll
        for (int c = 0; c < 8; c++) {
            int chunk = tid + c * 128;                  // 1024 chunks of 8
            int row = chunk >> 4, col0 = (chunk & 15) * 8;
            *(uint4*)&sK[row * 128 + col0] =
                *(const uint4*)&Kb[((size_t)b * NN + j + row) * DD + h * HDIM + col0];
            union { uint4 u; ushort s[8]; } vv;
            vv.u = *(const uint4*)&Vb[((size_t)b * NN + j + row) * DD + h * HDIM + col0];
            #pragma unroll
            for (int i = 0; i < 8; i++) sV[(col0 + i) * 64 + row] = vv.s[i];
        }
        __syncthreads();

        // S = Q @ K^T  (4 key tiles of 16, 4 K-steps each)
        floatx8 s[4] = {};
        #pragma unroll
        for (int t = 0; t < 4; t++) {
            #pragma unroll
            for (int kk = 0; kk < 4; kk++) {
                AFrag kb;
                const ushort* kp = &sK[(t * 16 + l16) * 128 + kk * 32];
                kb.h[0] = *(const bf16x8*)&kp[half * 8];
                kb.h[1] = *(const bf16x8*)&kp[16 + half * 8];
                s[t] = __builtin_amdgcn_wmma_f32_16x16x32_bf16(
                    false, qa[kk].v, false, kb.v, (short)0, s[t], false, false);
            }
        }

        // scale + mask, online softmax update
        floatx8 rm;
        #pragma unroll
        for (int r = 0; r < 8; r++) rm[r] = -3.0e38f;
        #pragma unroll
        for (int t = 0; t < 4; t++) {
            int jc = j + t * 16 + l16;
            bool tc = tgt[(size_t)b * NN + jc] != 0;
            #pragma unroll
            for (int r = 0; r < 8; r++) {
                int qi = q0 + r + 8 * half;
                float v = s[t][r] * inv_scale;
                if (tr[r] && tc && (qi != jc)) v = -3.0e38f;
                s[t][r] = v;
                rm[r] = fmaxf(rm[r], v);
            }
        }
        #pragma unroll
        for (int off = 1; off < 16; off <<= 1)
            #pragma unroll
            for (int r = 0; r < 8; r++)
                rm[r] = fmaxf(rm[r], __shfl_xor(rm[r], off));

        floatx8 newm, alpha, ps;
        #pragma unroll
        for (int r = 0; r < 8; r++) {
            newm[r]  = fmaxf(m[r], rm[r]);
            alpha[r] = __expf(m[r] - newm[r]);
            ps[r]    = 0.f;
        }
        #pragma unroll
        for (int t = 0; t < 4; t++)
            #pragma unroll
            for (int r = 0; r < 8; r++) {
                float sv = s[t][r];
                float p  = (sv < -1.0e37f) ? 0.f : __expf(sv - newm[r]);
                s[t][r] = p;
                ps[r]  += p;
            }
        #pragma unroll
        for (int off = 1; off < 16; off <<= 1)
            #pragma unroll
            for (int r = 0; r < 8; r++)
                ps[r] += __shfl_xor(ps[r], off);
        #pragma unroll
        for (int r = 0; r < 8; r++) {
            l[r] = l[r] * alpha[r] + ps[r];
            m[r] = newm[r];
        }
        #pragma unroll
        for (int t = 0; t < 8; t++)
            #pragma unroll
            for (int r = 0; r < 8; r++) o[t][r] *= alpha[r];

        // write P tile to per-wave LDS in A-matrix row-major [16][64] layout
        #pragma unroll
        for (int t = 0; t < 4; t++)
            #pragma unroll
            for (int r = 0; r < 8; r++)
                sP[wave][(r + 8 * half) * 64 + t * 16 + l16] = f2bf(s[t][r]);

        // O += P @ V  (2 K-steps of 32 keys, 8 hd-col tiles)
        #pragma unroll
        for (int kk2 = 0; kk2 < 2; kk2++) {
            AFrag pf;
            const ushort* pp = &sP[wave][l16 * 64 + kk2 * 32];
            pf.h[0] = *(const bf16x8*)&pp[half * 8];
            pf.h[1] = *(const bf16x8*)&pp[16 + half * 8];
            #pragma unroll
            for (int t8 = 0; t8 < 8; t8++) {
                AFrag vf;
                const ushort* vp = &sV[(t8 * 16 + l16) * 64 + kk2 * 32];
                vf.h[0] = *(const bf16x8*)&vp[half * 8];
                vf.h[1] = *(const bf16x8*)&vp[16 + half * 8];
                o[t8] = __builtin_amdgcn_wmma_f32_16x16x32_bf16(
                    false, pf.v, false, vf.v, (short)0, o[t8], false, false);
            }
        }
    }

    // normalize and store (bf16, feeds O-projection GEMM)
    #pragma unroll
    for (int t8 = 0; t8 < 8; t8++)
        #pragma unroll
        for (int r = 0; r < 8; r++) {
            int row = q0 + r + 8 * half;
            int col = h * HDIM + t8 * 16 + l16;
            Ob[((size_t)b * NN + row) * DD + col] = f2bf(o[t8][r] / l[r]);
        }
}

// ---------------------------------------------------------------------------
// out = LayerNorm(x + y); optional bf16 copy for next GEMM. One row per block.
// ---------------------------------------------------------------------------
__global__ __launch_bounds__(256)
void add_ln_kernel(const float* __restrict__ x, const float* __restrict__ y,
                   const float* __restrict__ g, const float* __restrict__ bt,
                   float* __restrict__ outf, ushort* __restrict__ outb) {
    const size_t base = (size_t)blockIdx.x * DD;
    const int tid = threadIdx.x, wave = tid >> 5, lane = tid & 31;
    float v[4], s = 0.f, s2 = 0.f;
    #pragma unroll
    for (int i = 0; i < 4; i++) {
        int c = tid + i * 256;
        v[i] = x[base + c] + y[base + c];
        s += v[i]; s2 += v[i] * v[i];
    }
    #pragma unroll
    for (int off = 1; off < 32; off <<= 1) {
        s  += __shfl_xor(s, off);
        s2 += __shfl_xor(s2, off);
    }
    __shared__ float rs[8], rs2[8];
    if (lane == 0) { rs[wave] = s; rs2[wave] = s2; }
    __syncthreads();
    float ts = 0.f, ts2 = 0.f;
    #pragma unroll
    for (int w = 0; w < 8; w++) { ts += rs[w]; ts2 += rs2[w]; }
    float mean = ts * (1.0f / DD);
    float var  = ts2 * (1.0f / DD) - mean * mean;
    float rstd = rsqrtf(var + 1e-5f);
    #pragma unroll
    for (int i = 0; i < 4; i++) {
        int c = tid + i * 256;
        float ov = (v[i] - mean) * rstd * g[c] + bt[c];
        outf[base + c] = ov;
        if (outb) outb[base + c] = f2bf(ov);
    }
}

// ---------------------------------------------------------------------------
// Host-side orchestration
// ---------------------------------------------------------------------------
extern "C" void kernel_launch(void* const* d_in, const int* in_sizes, int n_in,
                              void* d_out, int out_size, void* d_ws, size_t ws_size,
                              hipStream_t stream) {
    (void)in_sizes; (void)n_in; (void)out_size; (void)ws_size;

    const float* c_decor = (const float*)d_in[0];
    const unsigned char* is_target = (const unsigned char*)d_in[1];
    const float* Wq = (const float*)d_in[2];  const float* bq = (const float*)d_in[3];
    const float* Wk = (const float*)d_in[4];  const float* bk = (const float*)d_in[5];
    const float* Wv = (const float*)d_in[6];  const float* bv = (const float*)d_in[7];
    const float* Wo = (const float*)d_in[8];  const float* bo = (const float*)d_in[9];
    const float* ln1_g = (const float*)d_in[10]; const float* ln1_b = (const float*)d_in[11];
    const float* ln2_g = (const float*)d_in[12]; const float* ln2_b = (const float*)d_in[13];
    const float* W1 = (const float*)d_in[14]; const float* b1 = (const float*)d_in[15];
    const float* W2 = (const float*)d_in[16]; const float* b2 = (const float*)d_in[17];
    float* out = (float*)d_out;

    char* ws = (char*)d_ws;
    const size_t MB = 1024ull * 1024ull;
    ushort* WqT   = (ushort*)(ws + 0 * MB);       //  2 MB  [D][D] bf16 transposed
    ushort* WkT   = (ushort*)(ws + 2 * MB);
    ushort* WvT   = (ushort*)(ws + 4 * MB);
    ushort* WoT   = (ushort*)(ws + 6 * MB);
    ushort* W1T   = (ushort*)(ws + 8 * MB);       //  8 MB  [DFF][D]
    ushort* W2T   = (ushort*)(ws + 16 * MB);      //  8 MB  [D][DFF]
    ushort* x_bf  = (ushort*)(ws + 24 * MB);      // 16 MB  (later reused as attn out)
    ushort* Qbf   = (ushort*)(ws + 40 * MB);      // 16 MB
    ushort* Kbf   = (ushort*)(ws + 56 * MB);      // 16 MB
    ushort* Vbf   = (ushort*)(ws + 72 * MB);      // 16 MB
    float*  hout  = (float*) (ws + 88 * MB);      // 32 MB (later reused as ffn out)
    float*  cmidf = (float*) (ws + 120 * MB);     // 32 MB
    ushort* cmidb = (ushort*)(ws + 152 * MB);     // 16 MB
    ushort* ffnb  = (ushort*)(ws + 168 * MB);     // 64 MB
    ushort* attnb = x_bf;                         // reuse
    float*  ffnout = hout;                        // reuse

    const int nact = MTOT * DD;

    // 1) convert inputs/weights to bf16 (weights transposed)
    conv_bf16_kernel<<<(nact + 255) / 256, 256, 0, stream>>>(c_decor, x_bf, nact);
    transpose_bf16_kernel<<<dim3(DD / 32, DD / 32), 256, 0, stream>>>(Wq, WqT, DD, DD);
    transpose_bf16_kernel<<<dim3(DD / 32, DD / 32), 256, 0, stream>>>(Wk, WkT, DD, DD);
    transpose_bf16_kernel<<<dim3(DD / 32, DD / 32), 256, 0, stream>>>(Wv, WvT, DD, DD);
    transpose_bf16_kernel<<<dim3(DD / 32, DD / 32), 256, 0, stream>>>(Wo, WoT, DD, DD);
    transpose_bf16_kernel<<<dim3(DD / 32, DFF / 32), 256, 0, stream>>>(W1, W1T, DD, DFF);
    transpose_bf16_kernel<<<dim3(DFF / 32, DD / 32), 256, 0, stream>>>(W2, W2T, DFF, DD);

    // 2) QKV projections (bf16 out)
    dim3 gD(MTOT / 128, DD / 128);
    gemm_bf16_kernel<false, true><<<gD, 256, 0, stream>>>(x_bf, WqT, bq, Qbf, DD, DD);
    gemm_bf16_kernel<false, true><<<gD, 256, 0, stream>>>(x_bf, WkT, bk, Kbf, DD, DD);
    gemm_bf16_kernel<false, true><<<gD, 256, 0, stream>>>(x_bf, WvT, bv, Vbf, DD, DD);

    // 3) masked attention (writes attnb, reusing x_bf)
    attn_kernel<<<dim3(BB * HH, NN / 64), 128, 0, stream>>>(Qbf, Kbf, Vbf, is_target, attnb);

    // 4) output projection (fp32 out for residual)
    gemm_bf16_kernel<false, false><<<gD, 256, 0, stream>>>(attnb, WoT, bo, hout, DD, DD);

    // 5) c_mid = LN(c_decor + hout)
    add_ln_kernel<<<MTOT, 256, 0, stream>>>(c_decor, hout, ln1_g, ln1_b, cmidf, cmidb);

    // 6) FFN
    dim3 gF1(MTOT / 128, DFF / 128);
    gemm_bf16_kernel<true, true><<<gF1, 256, 0, stream>>>(cmidb, W1T, b1, ffnb, DFF, DD);
    gemm_bf16_kernel<false, false><<<gD, 256, 0, stream>>>(ffnb, W2T, b2, ffnout, DD, DFF);

    // 7) out = LN(c_mid + ffn)
    add_ln_kernel<<<MTOT, 256, 0, stream>>>(cmidf, ffnout, ln2_g, ln2_b, out, nullptr);
}
